// IPA_14035953123824
// MI455X (gfx1250) — compile-verified, matrix operand order checked by reading, not run on hardware
//
#include <hip/hip_runtime.h>
#include <hip/hip_bf16.h>
#include <math.h>

// IPA: B=2, N=512, C_S=128, C_HID=16, H=12, P=4, V=8
#define B_ 2
#define N_ 512
#define H_ 12
#define CS_ 128

typedef __attribute__((ext_vector_type(16))) _Float16 v16h;
typedef __attribute__((ext_vector_type(8)))  _Float16 h8;   // 16B, natural align 16
typedef __attribute__((ext_vector_type(8)))  float    v8f;
typedef __attribute__((ext_vector_type(4)))  float    f4v;  // 16B

// ---------------------------------------------------------------------------
// Generic 16x16-tile WMMA GEMM:  C[m, colOff+n] = A(MxK,f32) @ W(KxN,f32) + bias
// One wave per output tile. K must be a multiple of 32; A rows 16B-aligned.
// A-operand layout (16-bit A 16x32, ISA 7.12.2):
//   lane<16: elems 0..7 -> K=0..7,  8..15 -> K=16..23  (row M = lane)
//   lane>=16: elems 0..7 -> K=8..15, 8..15 -> K=24..31 (row M = lane-16)
// ---------------------------------------------------------------------------
__global__ __launch_bounds__(32) void gemm16_wmma(
    const float* __restrict__ A, int lda,
    const float* __restrict__ W, int ldb,
    const float* __restrict__ bias,
    float* __restrict__ C, int ldc, int colOff, int K)
{
    const int lane = threadIdx.x;
    const int nl   = lane & 15;
    const int hi   = lane >> 4;
    const int kOff = hi * 8;
    const int mBase = blockIdx.x * 16;
    const int nBase = blockIdx.y * 16;

    float bv = bias[nBase + nl];
    v8f acc;
#pragma unroll
    for (int r = 0; r < 8; ++r) acc[r] = bv;

    const float* aRow = A + (size_t)(mBase + nl) * lda;
    const float* bCol = W + (nBase + nl);

    for (int kb = 0; kb < K; kb += 32) {
        // A fragment: two contiguous 8-float (2x16B) runs -> f16
        f4v a0 = *(const f4v*)(aRow + kb + kOff);
        f4v a1 = *(const f4v*)(aRow + kb + kOff + 4);
        f4v a2 = *(const f4v*)(aRow + kb + kOff + 16);
        f4v a3 = *(const f4v*)(aRow + kb + kOff + 20);
        v16h af, bf;
#pragma unroll
        for (int e = 0; e < 4; ++e) {
            af[e]      = (_Float16)a0[e];
            af[4 + e]  = (_Float16)a1[e];
            af[8 + e]  = (_Float16)a2[e];
            af[12 + e] = (_Float16)a3[e];
        }
        // B fragment: column-strided (weights are KxN row-major)
#pragma unroll
        for (int e = 0; e < 8; ++e) {
            bf[e]     = (_Float16)bCol[(size_t)(kb + kOff + e) * ldb];
            bf[8 + e] = (_Float16)bCol[(size_t)(kb + kOff + 16 + e) * ldb];
        }
        acc = __builtin_amdgcn_wmma_f32_16x16x32_f16(
                  false, af, false, bf, (short)0, acc, false, false);
    }
    // C layout: VGPR r -> row mBase + r + hi*8, col nBase + nl
#pragma unroll
    for (int r = 0; r < 8; ++r) {
        int row = mBase + r + hi * 8;
        C[(size_t)row * ldc + colOff + nBase + nl] = acc[r];
    }
}

// ---------------------------------------------------------------------------
// Frame transform + pack into f16 WMMA-ready operands.
// Aaug[bh, n, 0:16]=q/sqrt(48), [16:28]=hw*qp(global), [28]=1, [29:32]=0
// Baug[bh, n, 0:16]=k,          [16:28]=kp, [28]=-0.5*hw*|kp|^2, [29:32]=0
// Vt  [bh, ch(48), n]  channel-major: ch 0:16=v, 16:40=vp(v-major xyz), 40:48=0
//   (channel-major makes the attention PV B-fragment loads contiguous b128)
// ---------------------------------------------------------------------------
__global__ __launch_bounds__(256) void transform_pack(
    const float* __restrict__ qkv,     // (B*N, 1152)
    const float* __restrict__ Rm,      // (B*N, 9)
    const float* __restrict__ tv,      // (B*N, 3)
    const float* __restrict__ head_w,  // (H)
    _Float16* __restrict__ Aaug,       // (B*H*N, 32)
    _Float16* __restrict__ Baug,       // (B*H*N, 32)
    _Float16* __restrict__ Vt)         // (B*H, 48, N)
{
    int idx = blockIdx.x * blockDim.x + threadIdx.x;
    if (idx >= B_ * N_ * H_) return;
    int h  = idx % H_;
    int bn = idx / H_;
    int n  = bn % N_;
    int b  = bn / N_;

    float R[3][3], t3[3];
#pragma unroll
    for (int i = 0; i < 3; ++i) {
#pragma unroll
        for (int j = 0; j < 3; ++j) R[i][j] = Rm[(size_t)bn * 9 + i * 3 + j];
        t3[i] = tv[(size_t)bn * 3 + i];
    }

    const float* row = qkv + (size_t)bn * 1152;
    // hw = softplus(w) * sqrt(1/(3*(P*9/2))) = softplus(w) * sqrt(1/54)
    float hw = log1pf(expf(head_w[h])) * 0.13608276348795434f;
    const float invs = 0.14433756729740643f;  // 1/sqrt(3*C_HID)

    int bh = b * H_ + h;
    _Float16* Ao = Aaug + ((size_t)bh * N_ + n) * 32;
    _Float16* Bo = Baug + ((size_t)bh * N_ + n) * 32;
    _Float16* Vo = Vt + (size_t)bh * 48 * N_ + n;   // stride N_ between channels

#pragma unroll
    for (int c = 0; c < 16; ++c) {
        Ao[c] = (_Float16)(row[h * 16 + c] * invs);
        Bo[c] = (_Float16)(row[192 + h * 32 + c]);
        Vo[(size_t)c * N_] = (_Float16)(row[192 + h * 32 + 16 + c]);
    }
    // query points: cols 576 + j*48 + (h*4+p)
#pragma unroll
    for (int p = 0; p < 4; ++p) {
        float r0 = row[576 +   0 + h * 4 + p];
        float r1 = row[576 +  48 + h * 4 + p];
        float r2 = row[576 +  96 + h * 4 + p];
#pragma unroll
        for (int i = 0; i < 3; ++i) {
            float v = R[i][0] * r0 + R[i][1] * r1 + R[i][2] * r2 + t3[i];
            Ao[16 + p * 3 + i] = (_Float16)(v * hw);
        }
    }
    Ao[28] = (_Float16)1.0f; Ao[29] = (_Float16)0.0f;
    Ao[30] = (_Float16)0.0f; Ao[31] = (_Float16)0.0f;

    // key/value points: cols 720 + j*144 + (h*12+p)
    float sqk = 0.0f;
#pragma unroll
    for (int p = 0; p < 12; ++p) {
        float r0 = row[720 +   0 + h * 12 + p];
        float r1 = row[720 + 144 + h * 12 + p];
        float r2 = row[720 + 288 + h * 12 + p];
#pragma unroll
        for (int i = 0; i < 3; ++i) {
            float v = R[i][0] * r0 + R[i][1] * r1 + R[i][2] * r2 + t3[i];
            if (p < 4) { Bo[16 + p * 3 + i] = (_Float16)v; sqk += v * v; }
            else       { Vo[(size_t)(16 + (p - 4) * 3 + i) * N_] = (_Float16)v; }
        }
    }
    Bo[28] = (_Float16)(-0.5f * hw * sqk);
    Bo[29] = (_Float16)0.0f; Bo[30] = (_Float16)0.0f; Bo[31] = (_Float16)0.0f;
#pragma unroll
    for (int c = 40; c < 48; ++c) Vo[(size_t)c * N_] = (_Float16)0.0f;
}

// ---------------------------------------------------------------------------
// Flash-style IPA attention. One wave per (b,h, 16-row i-tile).
// Per step: 2 logit WMMAs (K=32 augmented) over 32 j's, online softmax
// (row stats in VGPR r <-> row r/r+8; shfl_xor reductions over each half's
// 16 lanes), probs -> LDS -> A-layout fragment, 3 PV WMMAs over 48 channels.
// All fragment loads are contiguous 16B b128s.
// ---------------------------------------------------------------------------
__global__ __launch_bounds__(32) void ipa_attn(
    const _Float16* __restrict__ Aaug,    // (B*H*N, 32)
    const _Float16* __restrict__ Baug,    // (B*H*N, 32)
    const _Float16* __restrict__ Vt,      // (B*H, 48, N)
    const float* __restrict__ pair_mask,  // (B, N, N)
    float* __restrict__ oacc)             // (B*N, H, 48)
{
    __shared__ _Float16 pTile[16 * 32];

    const int lane = threadIdx.x;
    const int nl   = lane & 15;
    const int hi   = lane >> 4;
    const int kOff = hi * 8;
    const int iBase = blockIdx.x * 16;
    const int bh    = blockIdx.y;
    const int b     = bh / H_;
    const int h     = bh % H_;

    // A fragment (rows iBase..iBase+15, K=32 augmented)
    v16h af;
    {
        const _Float16* ar = Aaug + ((size_t)bh * N_ + iBase + nl) * 32 + kOff;
        h8 lo = *(const h8*)ar;
        h8 hi8 = *(const h8*)(ar + 16);
#pragma unroll
        for (int e = 0; e < 8; ++e) { af[e] = lo[e]; af[8 + e] = hi8[e]; }
    }

    v8f O0 = {0,0,0,0,0,0,0,0}, O1 = {0,0,0,0,0,0,0,0}, O2 = {0,0,0,0,0,0,0,0};
    float m[8], l[8];
#pragma unroll
    for (int r = 0; r < 8; ++r) { m[r] = -3.0e38f; l[r] = 0.0f; }

    const _Float16* vBase = Vt + (size_t)bh * 48 * N_;

    for (int jt = 0; jt < N_ / 32; ++jt) {
        const int j0 = jt * 32;

        const _Float16* b0p = Baug + ((size_t)bh * N_ + j0 + nl) * 32 + kOff;
        // prefetch next iteration's Baug rows (stream is L2-resident)
        if (jt + 1 < N_ / 32) __builtin_prefetch(b0p + 32 * 32, 0, 1);

        v16h bf0, bf1;
        {
            h8 x0 = *(const h8*)(b0p);
            h8 x1 = *(const h8*)(b0p + 16);
            h8 y0 = *(const h8*)(b0p + 16 * 32);
            h8 y1 = *(const h8*)(b0p + 16 * 32 + 16);
#pragma unroll
            for (int e = 0; e < 8; ++e) {
                bf0[e] = x0[e]; bf0[8 + e] = x1[e];
                bf1[e] = y0[e]; bf1[8 + e] = y1[e];
            }
        }
        v8f z = {0,0,0,0,0,0,0,0};
        v8f lg0 = __builtin_amdgcn_wmma_f32_16x16x32_f16(false, af, false, bf0,
                                                         (short)0, z, false, false);
        v8f lg1 = __builtin_amdgcn_wmma_f32_16x16x32_f16(false, af, false, bf1,
                                                         (short)0, z, false, false);
        // additive mask
#pragma unroll
        for (int r = 0; r < 8; ++r) {
            int row = iBase + r + hi * 8;
            const float* mr = pair_mask + ((size_t)b * N_ + row) * N_ + j0 + nl;
            lg0[r] += 100000.0f * (mr[0]  - 1.0f);
            lg1[r] += 100000.0f * (mr[16] - 1.0f);
        }
        // online softmax update
        float p0[8], p1[8];
#pragma unroll
        for (int r = 0; r < 8; ++r) {
            float tm = fmaxf(lg0[r], lg1[r]);
#pragma unroll
            for (int s = 1; s < 16; s <<= 1) tm = fmaxf(tm, __shfl_xor(tm, s, 32));
            float mn   = fmaxf(m[r], tm);
            float corr = __expf(m[r] - mn);
            l[r] *= corr;
            O0[r] *= corr; O1[r] *= corr; O2[r] *= corr;
            p0[r] = __expf(lg0[r] - mn);
            p1[r] = __expf(lg1[r] - mn);
            float rs = p0[r] + p1[r];
#pragma unroll
            for (int s = 1; s < 16; s <<= 1) rs += __shfl_xor(rs, s, 32);
            l[r] += rs;
            m[r] = mn;
        }
        // probs: C-layout -> LDS -> A-layout fragment (K=32 j-local)
#pragma unroll
        for (int r = 0; r < 8; ++r) {
            pTile[(r + hi * 8) * 32 + nl]      = (_Float16)p0[r];
            pTile[(r + hi * 8) * 32 + 16 + nl] = (_Float16)p1[r];
        }
        asm volatile("s_wait_dscnt 0" ::: "memory");
        v16h pf;
        {
            const _Float16* pr = pTile + nl * 32 + kOff;
            h8 lo = *(const h8*)pr;
            h8 hi8 = *(const h8*)(pr + 16);
#pragma unroll
            for (int e = 0; e < 8; ++e) { pf[e] = lo[e]; pf[8 + e] = hi8[e]; }
        }
        // PV: 3 channel tiles; channel-major Vt makes K contiguous -> 2xb128
#pragma unroll
        for (int t = 0; t < 3; ++t) {
            const _Float16* vr = vBase + (size_t)(t * 16 + nl) * N_ + j0 + kOff;
            h8 lo = *(const h8*)vr;
            h8 hi8 = *(const h8*)(vr + 16);
            v16h vf;
#pragma unroll
            for (int e = 0; e < 8; ++e) { vf[e] = lo[e]; vf[8 + e] = hi8[e]; }
            if      (t == 0) O0 = __builtin_amdgcn_wmma_f32_16x16x32_f16(false, pf, false, vf, (short)0, O0, false, false);
            else if (t == 1) O1 = __builtin_amdgcn_wmma_f32_16x16x32_f16(false, pf, false, vf, (short)0, O1, false, false);
            else             O2 = __builtin_amdgcn_wmma_f32_16x16x32_f16(false, pf, false, vf, (short)0, O2, false, false);
        }
        asm volatile("s_wait_dscnt 0" ::: "memory");  // WAR on pTile before next iter
    }
    // normalize + store (b, n, h, 48)
#pragma unroll
    for (int r = 0; r < 8; ++r) {
        float inv = 1.0f / l[r];
        int row = iBase + r + hi * 8;
        float* o = oacc + (((size_t)b * N_ + row) * H_ + h) * 48;
        o[nl]      = O0[r] * inv;
        o[16 + nl] = O1[r] * inv;
        o[32 + nl] = O2[r] * inv;
    }
}

// ---------------------------------------------------------------------------
// Inverse frame transform of o_pt, norm, and concat into (B*N, 576) f32.
// ---------------------------------------------------------------------------
__global__ __launch_bounds__(256) void finish_cat(
    const float* __restrict__ oacc,  // (B*N, H, 48)
    const float* __restrict__ Rm,
    const float* __restrict__ tv,
    float* __restrict__ cat)         // (B*N, 576)
{
    int idx = blockIdx.x * blockDim.x + threadIdx.x;
    if (idx >= B_ * N_ * H_) return;
    int h  = idx % H_;
    int bn = idx / H_;

    const float* o = oacc + ((size_t)bn * H_ + h) * 48;
    float R[3][3], t3[3];
#pragma unroll
    for (int i = 0; i < 3; ++i) {
#pragma unroll
        for (int j = 0; j < 3; ++j) R[i][j] = Rm[(size_t)bn * 9 + i * 3 + j];
        t3[i] = tv[(size_t)bn * 3 + i];
    }
    float* crow = cat + (size_t)bn * 576;
#pragma unroll
    for (int c = 0; c < 16; ++c) crow[h * 16 + c] = o[c];
#pragma unroll
    for (int v = 0; v < 8; ++v) {
        float g0 = o[16 + v * 3 + 0];
        float g1 = o[16 + v * 3 + 1];
        float g2 = o[16 + v * 3 + 2];
        // out_i = sum_j R[j][i]*g_j - t_i   (R^T rotation, then translate)
        float x = R[0][0] * g0 + R[1][0] * g1 + R[2][0] * g2 - t3[0];
        float y = R[0][1] * g0 + R[1][1] * g1 + R[2][1] * g2 - t3[1];
        float zc= R[0][2] * g0 + R[1][2] * g1 + R[2][2] * g2 - t3[2];
        float nm = sqrtf(x * x + y * y + zc * zc + 1e-8f);
        crow[192 + h * 8 + v] = x;
        crow[288 + h * 8 + v] = y;
        crow[384 + h * 8 + v] = zc;
        crow[480 + h * 8 + v] = nm;
    }
}

// ---------------------------------------------------------------------------
extern "C" void kernel_launch(void* const* d_in, const int* in_sizes, int n_in,
                              void* d_out, int out_size, void* d_ws, size_t ws_size,
                              hipStream_t stream) {
    const float* s         = (const float*)d_in[0];
    const float* R         = (const float*)d_in[1];
    const float* t         = (const float*)d_in[2];
    const float* pair_mask = (const float*)d_in[3];
    const float* Wq        = (const float*)d_in[4];
    const float* bq        = (const float*)d_in[5];
    const float* Wkv       = (const float*)d_in[6];
    const float* bkv       = (const float*)d_in[7];
    const float* Wqp       = (const float*)d_in[8];
    const float* bqp       = (const float*)d_in[9];
    const float* Wkvp      = (const float*)d_in[10];
    const float* bkvp      = (const float*)d_in[11];
    const float* head_w    = (const float*)d_in[12];
    const float* Wout      = (const float*)d_in[13];
    const float* bout      = (const float*)d_in[14];

    const int M   = B_ * N_;        // 1024
    const int BHN = B_ * H_ * N_;   // 12288

    char* p = (char*)d_ws;
    float*     qkv  = (float*)p;     p += (size_t)M * 1152 * sizeof(float);
    _Float16*  Aaug = (_Float16*)p;  p += (size_t)BHN * 32 * sizeof(_Float16);
    _Float16*  Baug = (_Float16*)p;  p += (size_t)BHN * 32 * sizeof(_Float16);
    _Float16*  Vt   = (_Float16*)p;  p += (size_t)BHN * 48 * sizeof(_Float16);
    float*     oacc = (float*)p;     p += (size_t)BHN * 48 * sizeof(float);
    float*     cat  = (float*)p;     p += (size_t)M * 576 * sizeof(float);

    // 1) projections: qkv = [s@Wq | s@Wkv | s@Wqp | s@Wkvp] + biases
    gemm16_wmma<<<dim3(M / 16, 192 / 16), 32, 0, stream>>>(s, CS_, Wq,   192, bq,   qkv, 1152,   0, CS_);
    gemm16_wmma<<<dim3(M / 16, 384 / 16), 32, 0, stream>>>(s, CS_, Wkv,  384, bkv,  qkv, 1152, 192, CS_);
    gemm16_wmma<<<dim3(M / 16, 144 / 16), 32, 0, stream>>>(s, CS_, Wqp,  144, bqp,  qkv, 1152, 576, CS_);
    gemm16_wmma<<<dim3(M / 16, 432 / 16), 32, 0, stream>>>(s, CS_, Wkvp, 432, bkvp, qkv, 1152, 720, CS_);

    // 2) frame transform + f16 packing (Vt channel-major for b128 fragment loads)
    transform_pack<<<BHN / 256, 256, 0, stream>>>(qkv, R, t, head_w, Aaug, Baug, Vt);

    // 3) flash-IPA attention
    ipa_attn<<<dim3(N_ / 16, B_ * H_), 32, 0, stream>>>(Aaug, Baug, Vt, pair_mask, oacc);

    // 4) inverse transform + concat
    finish_cat<<<BHN / 256, 256, 0, stream>>>(oacc, R, t, cat);

    // 5) output projection into d_out
    gemm16_wmma<<<dim3(M / 16, CS_ / 16), 32, 0, stream>>>(cat, 576, Wout, CS_, bout,
                                                           (float*)d_out, CS_, 0, 576);
}
